// _ResGCN_21440476741915
// MI455X (gfx1250) — matrix operand order repaired and checked by reading, not compiled, not run on hardware
//
#include <hip/hip_runtime.h>

typedef __attribute__((ext_vector_type(16))) _Float16 v16h;
typedef __attribute__((ext_vector_type(8)))  float    v8f;

#define N_NODES 20000
#define N_EDGES 160000
#define NB      16
#define NC      16
#define NH      64
#define NODE_F  140
#define CFG_F   18
#define EMBD    32
#define R_ROWS  (N_NODES * NC)   /* 320000 rows */
#define KP1     192              /* 190 padded to 192 */
#define SLOPE   0.2f

static __device__ __forceinline__ float leaky(float v) {
    return v > 0.f ? v : SLOPE * v;
}

// ---------------------------------------------------------------- degree
__global__ void k_deg_init(float* __restrict__ deg, int n) {
    int i = blockIdx.x * blockDim.x + threadIdx.x;
    if (i < n) deg[i] = 1.0f;
}
__global__ void k_deg_acc(const int* __restrict__ src, float* __restrict__ deg, int e) {
    int i = blockIdx.x * blockDim.x + threadIdx.x;
    if (i < e) atomicAdd(&deg[src[i]], 1.0f);
}
__global__ void k_deg_inv(float* __restrict__ deg, int n) {
    int i = blockIdx.x * blockDim.x + threadIdx.x;
    if (i < n) deg[i] = 1.0f / deg[i];
}

// ------------------------------------------------- stage concat input (f16)
__global__ void k_stage(const int* __restrict__ op_ids,
                        const float* __restrict__ nf,
                        const float* __restrict__ cf,
                        const float* __restrict__ emb,
                        _Float16* __restrict__ A, int total) {
    int idx = blockIdx.x * blockDim.x + threadIdx.x;
    if (idx >= total) return;
    int k    = idx % KP1;
    int row  = idx / KP1;
    int node = row >> 4;
    float v;
    if (k < EMBD)               v = emb[(size_t)op_ids[node] * EMBD + k];
    else if (k < EMBD + NODE_F) v = nf[(size_t)node * NODE_F + (k - EMBD)];
    else if (k < 190)           v = cf[(size_t)row * CFG_F + (k - EMBD - NODE_F)];
    else                        v = 0.f;
    A[idx] = (_Float16)v;
}

// ---------------------------------------- pack weight (KxN f32) to B-fragments
// layout: Bp[(((kt*4+nt)*32 + lane)*16 + e)] = W[kt*32 + 16*(lane>=16) + e][nt*16 + (lane&15)]
__global__ void k_pack(const float* __restrict__ W, _Float16* __restrict__ Bp,
                       int K, int KPAD) {
    int idx = blockIdx.x * blockDim.x + threadIdx.x;
    if (idx >= KPAD * 64) return;
    int e    = idx & 15;
    int lane = (idx >> 4) & 31;
    int nt   = (idx >> 9) & 3;
    int kt   = idx >> 11;
    int n = nt * 16 + (lane & 15);
    int k = kt * 32 + (lane >> 4) * 16 + e;
    float v = (k < K) ? W[(size_t)k * 64 + n] : 0.f;
    Bp[idx] = (_Float16)v;
}

// ---------------------------------------------------------------- WMMA GEMM
// One wave computes a 16x64 output tile. A: R x KPAD f16 row-major.
template<int KPAD, bool ACT, bool OUTH, bool RESID>
__global__ void k_gemm(const _Float16* __restrict__ A,
                       const _Float16* __restrict__ Bp,
                       const float* __restrict__ bias,
                       float* __restrict__ outF,
                       _Float16* __restrict__ outH, int R) {
    int wave = (int)((blockIdx.x * blockDim.x + threadIdx.x) >> 5);
    int lane = threadIdx.x & 31;
    int rowBase = wave * 16;
    if (rowBase >= R) return;
    int hi = lane >> 4;
    int m  = lane & 15;
    const _Float16* Arow = A + (size_t)(rowBase + m) * KPAD;

    v8f acc0 = {}, acc1 = {}, acc2 = {}, acc3 = {};
    union Frag { v16h v; uint4 u[2]; };

    #pragma unroll
    for (int kt = 0; kt < KPAD / 32; ++kt) {
        int k0 = kt * 32 + hi * 16;
        Frag a;
        a.u[0] = *(const uint4*)(Arow + k0);
        a.u[1] = *(const uint4*)(Arow + k0 + 8);
        const _Float16* bp = Bp + ((size_t)(kt * 4) * 32 + lane) * 16;
        Frag b0, b1, b2, b3;
        b0.u[0] = *(const uint4*)(bp);           b0.u[1] = *(const uint4*)(bp + 8);
        b1.u[0] = *(const uint4*)(bp + 512);     b1.u[1] = *(const uint4*)(bp + 520);
        b2.u[0] = *(const uint4*)(bp + 1024);    b2.u[1] = *(const uint4*)(bp + 1032);
        b3.u[0] = *(const uint4*)(bp + 1536);    b3.u[1] = *(const uint4*)(bp + 1544);
        acc0 = __builtin_amdgcn_wmma_f32_16x16x32_f16(false, a.v, false, b0.v, (short)0, acc0, false, false);
        acc1 = __builtin_amdgcn_wmma_f32_16x16x32_f16(false, a.v, false, b1.v, (short)0, acc1, false, false);
        acc2 = __builtin_amdgcn_wmma_f32_16x16x32_f16(false, a.v, false, b2.v, (short)0, acc2, false, false);
        acc3 = __builtin_amdgcn_wmma_f32_16x16x32_f16(false, a.v, false, b3.v, (short)0, acc3, false, false);
    }

    v8f accs[4] = {acc0, acc1, acc2, acc3};
    #pragma unroll
    for (int nt = 0; nt < 4; ++nt) {
        int col = nt * 16 + m;
        float bv = bias[col];
        #pragma unroll
        for (int j = 0; j < 8; ++j) {
            int r = rowBase + hi * 8 + j;
            size_t o = (size_t)r * 64 + col;
            float v = accs[nt][j] + bv;
            if (RESID) v += outF[o];
            if (ACT)   v = leaky(v);
            if (OUTH)  outH[o] = (_Float16)v;
            else       outF[o] = v;
        }
    }
}

// ---------------------------------------------------------------- act
__global__ void k_act(const float* __restrict__ x, float* __restrict__ y, int total4) {
    int i = blockIdx.x * blockDim.x + threadIdx.x;
    if (i >= total4) return;
    const float4 v = *(const float4*)(x + (size_t)i * 4);
    float4 o;
    o.x = leaky(v.x); o.y = leaky(v.y); o.z = leaky(v.z); o.w = leaky(v.w);
    *(float4*)(y + (size_t)i * 4) = o;
}

// --------------------------------------------------- edge scatter (both dirs)
// agg[dst] += y[src]/deg[src] ; agg[src] += y[dst]/deg[src]
__global__ void k_edge(const int* __restrict__ src, const int* __restrict__ dst,
                       const float* __restrict__ invdeg, const float* __restrict__ y,
                       float* __restrict__ agg, int total) {
    int idx = blockIdx.x * blockDim.x + threadIdx.x;
    if (idx >= total) return;
    int e  = idx >> 8;
    int fo = (idx & 255) << 2;
    int s = src[e], d = dst[e];
    float inv = invdeg[s];
    const float4 ys = *(const float4*)(y + (size_t)s * 1024 + fo);
    const float4 yd = *(const float4*)(y + (size_t)d * 1024 + fo);
    float* ad = agg + (size_t)d * 1024 + fo;
    float* as = agg + (size_t)s * 1024 + fo;
    atomicAdd(ad + 0, ys.x * inv); atomicAdd(ad + 1, ys.y * inv);
    atomicAdd(ad + 2, ys.z * inv); atomicAdd(ad + 3, ys.w * inv);
    atomicAdd(as + 0, yd.x * inv); atomicAdd(as + 1, yd.y * inv);
    atomicAdd(as + 2, yd.z * inv); atomicAdd(as + 3, yd.w * inv);
}

// --------------------------------------------- combine: z = agg + y*(1+2/deg)
__global__ void k_combine(const float* __restrict__ agg, const float* __restrict__ y,
                          const float* __restrict__ invdeg, _Float16* __restrict__ z16,
                          int total4) {
    int i = blockIdx.x * blockDim.x + threadIdx.x;
    if (i >= total4) return;
    size_t f = (size_t)i * 4;
    int node = (int)(f >> 10);
    float s = 1.f + 2.f * invdeg[node];
    const float4 a  = *(const float4*)(agg + f);
    const float4 yy = *(const float4*)(y + f);
    z16[f + 0] = (_Float16)(a.x + yy.x * s);
    z16[f + 1] = (_Float16)(a.y + yy.y * s);
    z16[f + 2] = (_Float16)(a.z + yy.z * s);
    z16[f + 3] = (_Float16)(a.w + yy.w * s);
}

// ---------------------------------------------------------------- pool
__global__ void k_pool(const float* __restrict__ x, const int* __restrict__ gid,
                       float* __restrict__ pooled, int n) {
    int f  = blockIdx.x * blockDim.x + threadIdx.x;   // 0..1023
    int n0 = blockIdx.y * 250;
    int n1 = n0 + 250; if (n1 > n) n1 = n;
    if (n0 >= n) return;
    float sum = 0.f;
    int cur = gid[n0];
    for (int i = n0; i < n1; ++i) {
        int g = gid[i];
        if (g != cur) { atomicAdd(&pooled[(size_t)cur * 1024 + f], sum); sum = 0.f; cur = g; }
        sum += leaky(x[(size_t)i * 1024 + f]);
    }
    atomicAdd(&pooled[(size_t)cur * 1024 + f], sum);
}

// ---------------------------------------------------------------- postnet
__global__ void k_post(const float* __restrict__ pooled,
                       const float* __restrict__ W0, const float* __restrict__ b0,
                       const float* __restrict__ W1, const float* __restrict__ b1,
                       float* __restrict__ out) {
    int r = threadIdx.x;  // 256 rows = (B,C)
    float h[64];
    #pragma unroll 4
    for (int j = 0; j < 64; ++j) {
        float s = b0[j];
        for (int k = 0; k < 64; ++k) s += pooled[(size_t)r * 64 + k] * W0[(size_t)k * 64 + j];
        h[j] = s;
    }
    float o = b1[0];
    for (int j = 0; j < 64; ++j) o += leaky(h[j]) * W1[j];
    out[r] = o;
}

static inline int cdiv(int a, int b) { return (a + b - 1) / b; }

extern "C" void kernel_launch(void* const* d_in, const int* in_sizes, int n_in,
                              void* d_out, int out_size, void* d_ws, size_t ws_size,
                              hipStream_t stream) {
    (void)n_in; (void)out_size; (void)ws_size;
    const int*   op_ids = (const int*)  d_in[0];
    const float* nf     = (const float*)d_in[1];
    const float* cf     = (const float*)d_in[2];
    const int*   src    = (const int*)  d_in[3];
    const int*   dst    = (const int*)  d_in[4];
    const int*   gid    = (const int*)  d_in[5];

    // ---- parameter layout detection (insertion order vs sorted tree-flatten)
    const float *emb, *preW0, *preB0, *preW1, *preB1;
    const float *gcW[6], *gcB[6];
    const float *postW0, *postB0, *postW1, *postB1;
    int p = 6;
    emb = (const float*)d_in[p++];                       // 120*32
    if (in_sizes[p] == (190 * 64)) {
        // insertion order: prenet, gc, postnet
        preW0 = (const float*)d_in[p++]; preB0 = (const float*)d_in[p++];
        preW1 = (const float*)d_in[p++]; preB1 = (const float*)d_in[p++];
        for (int i = 0; i < 6; ++i) { gcW[i] = (const float*)d_in[p++]; gcB[i] = (const float*)d_in[p++]; }
        postW0 = (const float*)d_in[p++]; postB0 = (const float*)d_in[p++];
        postW1 = (const float*)d_in[p++]; postB1 = (const float*)d_in[p++];
    } else {
        // sorted keys: gc, postnet, prenet
        for (int i = 0; i < 6; ++i) { gcW[i] = (const float*)d_in[p++]; gcB[i] = (const float*)d_in[p++]; }
        postW0 = (const float*)d_in[p++]; postB0 = (const float*)d_in[p++];
        postW1 = (const float*)d_in[p++]; postB1 = (const float*)d_in[p++];
        preW0 = (const float*)d_in[p++]; preB0 = (const float*)d_in[p++];
        preW1 = (const float*)d_in[p++]; preB1 = (const float*)d_in[p++];
    }

    // ---- workspace carve-out
    char* ws = (char*)d_ws;
    size_t off = 0;
    auto carve = [&](size_t bytes) -> char* {
        char* q = ws + off;
        off += (bytes + 255) & ~(size_t)255;
        return q;
    };
    float*    deg    = (float*)   carve((size_t)N_NODES * 4);
    float*    x      = (float*)   carve((size_t)R_ROWS * NH * 4);
    float*    y      = (float*)   carve((size_t)R_ROWS * NH * 4);
    float*    agg    = (float*)   carve((size_t)R_ROWS * NH * 4);
    float*    pooled = (float*)   carve((size_t)NB * NC * NH * 4);
    _Float16* wp     = (_Float16*)carve((size_t)8 * 12288 * 2);   // 8 packed-weight slots
    _Float16* a16    = (_Float16*)carve((size_t)R_ROWS * KP1 * 2);
    _Float16* b16    = (_Float16*)carve((size_t)R_ROWS * NH * 2);

    float* out = (float*)d_out;

    // ---- degree (stores 1/deg)
    k_deg_init<<<cdiv(N_NODES, 256), 256, 0, stream>>>(deg, N_NODES);
    k_deg_acc <<<cdiv(N_EDGES, 256), 256, 0, stream>>>(src, deg, N_EDGES);
    k_deg_inv <<<cdiv(N_NODES, 256), 256, 0, stream>>>(deg, N_NODES);

    // ---- pack weights into WMMA B-fragment layout
    k_pack<<<cdiv(KP1 * 64, 256), 256, 0, stream>>>(preW0, wp + 0 * 12288, 190, KP1);
    k_pack<<<cdiv(64 * 64, 256), 256, 0, stream>>>(preW1, wp + 1 * 12288, 64, 64);
    for (int i = 0; i < 6; ++i)
        k_pack<<<cdiv(64 * 64, 256), 256, 0, stream>>>(gcW[i], wp + (2 + i) * 12288, 64, 64);

    // ---- stage concat input (f16, R x 192)
    k_stage<<<cdiv(R_ROWS * KP1, 256), 256, 0, stream>>>(op_ids, nf, cf, emb, a16, R_ROWS * KP1);

    // ---- prenet: (R x 192)@W0 +b0 -> act -> f16 ; @W1 +b1 -> f32 x
    const int gemmBlocks = cdiv(R_ROWS / 16, 8);  // 8 waves/block, 16 rows/wave
    k_gemm<KP1, true,  true,  false><<<gemmBlocks, 256, 0, stream>>>(a16, wp + 0 * 12288, preB0, nullptr, b16, R_ROWS);
    k_gemm<64,  false, false, false><<<gemmBlocks, 256, 0, stream>>>(b16, wp + 1 * 12288, preB1, x, nullptr, R_ROWS);

    // ---- 3 GNN layers
    const int e4  = N_EDGES * 256;             // threads for edge kernel (4 floats each)
    const int t4  = (R_ROWS * NH) / 4;         // vec4 elementwise threads
    for (int l = 0; l < 3; ++l) {
        k_act<<<cdiv(t4, 256), 256, 0, stream>>>(x, y, t4);
        hipMemsetAsync(agg, 0, (size_t)R_ROWS * NH * 4, stream);
        k_edge<<<cdiv(e4, 256), 256, 0, stream>>>(src, dst, deg, y, agg, e4);
        k_combine<<<cdiv(t4, 256), 256, 0, stream>>>(agg, y, deg, a16, t4);
        k_gemm<64, true,  true,  false><<<gemmBlocks, 256, 0, stream>>>(a16, wp + (2 + 2 * l) * 12288, gcB[2 * l],     nullptr, b16, R_ROWS);
        k_gemm<64, false, false, true ><<<gemmBlocks, 256, 0, stream>>>(b16, wp + (3 + 2 * l) * 12288, gcB[2 * l + 1], x,       nullptr, R_ROWS);
    }

    // ---- final act + pooled segment-sum + postnet
    hipMemsetAsync(pooled, 0, (size_t)NB * NC * NH * 4, stream);
    {
        dim3 g(4, cdiv(N_NODES, 250));
        k_pool<<<g, 256, 0, stream>>>(x, gid, pooled, N_NODES);
    }
    k_post<<<1, 256, 0, stream>>>(pooled, postW0, postB0, postW1, postB1, out);
}
